// FluidAMG_10548439679332
// MI455X (gfx1250) — compile-verified, matrix operand order checked by reading, not run on hardware
//
#include <hip/hip_runtime.h>
#include <math.h>

typedef __attribute__((ext_vector_type(16))) _Float16 v16h;
typedef __attribute__((ext_vector_type(8)))  float    v8f;

// ---------------- wave32 helpers ----------------
__device__ __forceinline__ float wsum(float v){
#pragma unroll
  for (int o = 16; o > 0; o >>= 1) v += __shfl_xor(v, o, 32);
  return v;
}
__device__ __forceinline__ float wmaxv(float v){
#pragma unroll
  for (int o = 16; o > 0; o >>= 1) v = fmaxf(v, __shfl_xor(v, o, 32));
  return v;
}
__device__ __forceinline__ float gelu_f(float x){
  return 0.5f * x * (1.0f + erff(x * 0.7071067811865475f));
}
__device__ __forceinline__ void atomicMaxF(float* a, float v){
  if (v >= 0.f) atomicMax((int*)a, __float_as_int(v));
  else          atomicMin((unsigned int*)a, __float_as_uint(v));
}

// ---------------- weight pre-pack into WMMA B-fragment layout ----------------
// B[K,N] f32  ->  Bp dwords laid out [(tileN*Ksteps + ks)*32 + lane]*8 + v,
// dword v holds f16 pair (k, k+1) for this lane's column, per the ISA 16-bit layout.
__global__ void pack_b_f16(const float* __restrict__ B, unsigned* __restrict__ Bp,
                           int K, int N)
{
  int i = blockIdx.x * blockDim.x + threadIdx.x;
  int total = (N >> 4) * (K >> 5) * 256;
  if (i >= total) return;
  int v      = i & 7;
  int lane   = (i >> 3) & 31;
  int rest   = i >> 8;
  int ksteps = K >> 5;
  int ks = rest % ksteps;
  int t  = rest / ksteps;
  int khalf = (lane >> 4) * 8;
  int k   = ks*32 + ((v < 4) ? (2*v + khalf) : (16 + 2*(v-4) + khalf));
  int col = t*16 + (lane & 15);
  union { _Float16 h[2]; unsigned u; } cv;
  cv.h[0] = (_Float16)B[(size_t)k    *N + col];
  cv.h[1] = (_Float16)B[(size_t)(k+1)*N + col];
  Bp[i] = cv.u;
}

// ---------------- generic WMMA GEMM ----------------
// C[M,N] = epi(A[M,K] @ B[K,N] + bias[N] (+ res[M,N])), B pre-packed f16 fragments.
// Each wave computes a 16x64 output strip (4 N-tiles): the A fragment is loaded and
// converted once per K-step and fed to 4 independent WMMAs. 4 waves/block = 64 rows.
// Requires N % 64 == 0 (all GEMMs here have N in {128,256,512}). K is compile-time.
template<int EPI, int K>   // EPI: 0 = none, 1 = exact GELU
__global__ void gemm_wmma(const float* __restrict__ A, const unsigned* __restrict__ Bp,
                          const float* __restrict__ bias, const float* __restrict__ res,
                          float* __restrict__ C, int M, int N)
{
  constexpr int KS = K / 32;                 // K-steps
  int lane   = threadIdx.x & 31;
  int wave   = threadIdx.x >> 5;
  int tileN0 = blockIdx.x * 4;               // first of 4 consecutive N-tiles
  int tileM  = blockIdx.y * 4 + wave;
  if (tileM * 16 >= M) return;               // uniform per wave; EXEC stays all-ones
  int m0 = tileM * 16;
  int row = m0 + (lane & 15);
  if (row > M - 1) row = M - 1;              // clamp: junk rows never stored
  int coll  = lane & 15;
  int khalf = (lane >> 4) * 8;               // lanes 16-31 hold K+8 pairs
  const float*    arow = A  + (size_t)row * K;
  const unsigned* bb   = Bp + (size_t)tileN0 * KS * 256 + lane * 8;

  v8f acc[4] = {};
#pragma unroll
  for (int ks = 0; ks < KS; ++ks) {
    int k0 = ks * 32;
    float4 a0 = *(const float4*)(arow + k0 + khalf);
    float4 a1 = *(const float4*)(arow + k0 + khalf + 4);
    float4 a2 = *(const float4*)(arow + k0 + 16 + khalf);
    float4 a3 = *(const float4*)(arow + k0 + 16 + khalf + 4);
    v16h af;
    af[0]  = (_Float16)a0.x; af[1]  = (_Float16)a0.y;
    af[2]  = (_Float16)a0.z; af[3]  = (_Float16)a0.w;
    af[4]  = (_Float16)a1.x; af[5]  = (_Float16)a1.y;
    af[6]  = (_Float16)a1.z; af[7]  = (_Float16)a1.w;
    af[8]  = (_Float16)a2.x; af[9]  = (_Float16)a2.y;
    af[10] = (_Float16)a2.z; af[11] = (_Float16)a2.w;
    af[12] = (_Float16)a3.x; af[13] = (_Float16)a3.y;
    af[14] = (_Float16)a3.z; af[15] = (_Float16)a3.w;
    const unsigned* bs = bb + (size_t)ks * 256;
#pragma unroll
    for (int t = 0; t < 4; ++t) {
      v16h bf = *(const v16h*)(bs + (size_t)t * KS * 256);
      acc[t] = __builtin_amdgcn_wmma_f32_16x16x32_f16(false, af, false, bf,
                                                      (short)0, acc[t], false, false);
    }
  }
  int rbase = m0 + ((lane >> 4) ? 8 : 0);    // C layout: VGPR r -> M = r (+8 hi lanes)
#pragma unroll
  for (int t = 0; t < 4; ++t) {
    int col = (tileN0 + t) * 16 + coll;
    float bv = bias ? bias[col] : 0.f;
#pragma unroll
    for (int r = 0; r < 8; ++r) {
      int m = rbase + r;
      if (m < M) {
        float v = acc[t][r] + bv;
        if (res) v += res[(size_t)m*N + col];
        if (EPI == 1) v = gelu_f(v);
        C[(size_t)m*N + col] = v;
      }
    }
  }
}

// ---------------- elementwise / LN ----------------
__global__ void fillf(float* __restrict__ p, float v, size_t n){
  size_t i = (size_t)blockIdx.x * blockDim.x + threadIdx.x;
  if (i < n) p[i] = v;
}

// one wave per 128-wide row: y = LN(x (+res) (+eb)) * g + b
__global__ void ln_rows(const float* __restrict__ x, const float* __restrict__ res,
                        const float* __restrict__ eb, const float* __restrict__ g,
                        const float* __restrict__ b, float* __restrict__ y, int n)
{
  int row  = (blockIdx.x * blockDim.x + threadIdx.x) >> 5;
  int lane = threadIdx.x & 31;
  if (row >= n) return;
  const float* xr = x + (size_t)row * 128;
  float v[4]; float s = 0.f;
#pragma unroll
  for (int i = 0; i < 4; ++i) {
    int d = lane + 32*i;
    float t = xr[d];
    if (res) t += res[(size_t)row*128 + d];
    if (eb)  t += eb[d];
    v[i] = t; s += t;
  }
  s = wsum(s);
  float mean = s * (1.f/128.f);
  float q = 0.f;
#pragma unroll
  for (int i = 0; i < 4; ++i) { float dv = v[i] - mean; q += dv*dv; }
  q = wsum(q);
  float inv = rsqrtf(q * (1.f/128.f) + 1e-5f);
#pragma unroll
  for (int i = 0; i < 4; ++i) {
    int d = lane + 32*i;
    y[(size_t)row*128 + d] = (v[i] - mean) * inv * g[d] + b[d];
  }
}

// ---------------- encoder / decoder tails ----------------
__global__ void enc_first(const float* __restrict__ x, const float* __restrict__ w1,
                          const float* __restrict__ b1, float* __restrict__ t, int n)
{
  int i = blockIdx.x * blockDim.x + threadIdx.x;
  if (i >= n * 128) return;
  int nd = i >> 7, d = i & 127;
  float s = b1[d];
#pragma unroll
  for (int j = 0; j < 9; ++j) s += x[nd*9 + j] * w1[j*128 + d];
  t[i] = gelu_f(s);
}

__global__ void dec_last(const float* __restrict__ u, const float* __restrict__ w2,
                         const float* __restrict__ b2, float* __restrict__ out, int n)
{
  int i = blockIdx.x * blockDim.x + threadIdx.x;
  if (i >= n * 4) return;
  int nd = i >> 2, c = i & 3;
  float s = b2[c];
  const float* ur = u + (size_t)nd * 128;
#pragma unroll 8
  for (int d = 0; d < 128; ++d) s += ur[d] * w2[d*4 + c];
  out[i] = s;
}

// ---------------- GATv2 edge kernels ----------------
// virtual edges [E, E+n) are self-loops
__global__ void gat_logits(const int* __restrict__ src, const int* __restrict__ dst,
                           int E, int n, const float* __restrict__ xl,
                           const float* __restrict__ xr, const float* __restrict__ att,
                           float* __restrict__ logit, float* __restrict__ mb)
{
  int gw   = (blockIdx.x * blockDim.x + threadIdx.x) >> 5;
  int lane = threadIdx.x & 31;
  if (gw >= E + n) return;
  int s = (gw < E) ? src[gw] : (gw - E);
  int d = (gw < E) ? dst[gw] : (gw - E);
  const float* xls = xl + (size_t)s * 512;
  const float* xrd = xr + (size_t)d * 512;
#pragma unroll
  for (int hh = 0; hh < 4; ++hh) {
    float acc = 0.f;
#pragma unroll
    for (int cc = 0; cc < 4; ++cc) {
      int dd = hh*128 + lane + 32*cc;
      float e = xls[dd] + xrd[dd];
      e = (e > 0.f) ? e : 0.2f * e;       // leaky_relu(0.2)
      acc += e * att[dd];
    }
    acc = wsum(acc);
    if (lane == 0) {
      logit[(size_t)gw*4 + hh] = acc;
      atomicMaxF(&mb[(size_t)d*4 + hh], acc);
    }
  }
}

__global__ void gat_expsum(const int* __restrict__ dst, int E, int n,
                           float* __restrict__ logit, const float* __restrict__ mb,
                           float* __restrict__ zb)
{
  int i = blockIdx.x * blockDim.x + threadIdx.x;
  if (i >= (E + n) * 4) return;
  int e = i >> 2, hh = i & 3;
  int d = (e < E) ? dst[e] : (e - E);
  float a = expf(logit[i] - mb[(size_t)d*4 + hh]);
  logit[i] = a;
  atomicAdd(&zb[(size_t)d*4 + hh], a);
}

__global__ void gat_scatter(const int* __restrict__ src, const int* __restrict__ dst,
                            int E, int n, const float* __restrict__ xl,
                            const float* __restrict__ logit, const float* __restrict__ zb,
                            float* __restrict__ out)
{
  int gw   = (blockIdx.x * blockDim.x + threadIdx.x) >> 5;
  int lane = threadIdx.x & 31;
  if (gw >= E + n) return;
  int s = (gw < E) ? src[gw] : (gw - E);
  int d = (gw < E) ? dst[gw] : (gw - E);
  float wgt[4];
#pragma unroll
  for (int hh = 0; hh < 4; ++hh)
    wgt[hh] = logit[(size_t)gw*4 + hh] / zb[(size_t)d*4 + hh];
  const float* xls = xl + (size_t)s * 512;
#pragma unroll
  for (int cc = 0; cc < 4; ++cc) {
    int dd = lane + 32*cc;
    float acc = 0.f;
#pragma unroll
    for (int hh = 0; hh < 4; ++hh) acc += wgt[hh] * xls[hh*128 + dd];
    atomicAdd(&out[(size_t)d*128 + dd], 0.25f * acc);   // mean over 4 heads
  }
}

// ---------------- physics attention ----------------
__global__ void attn_in_scores(const float* __restrict__ Qt, const float* __restrict__ Kh,
                               float* __restrict__ s, int n)
{
  int i = blockIdx.x * blockDim.x + threadIdx.x;
  if (i >= 128 * n) return;
  int k = i % n; int qh = i / n; int hh = qh >> 5; int q = qh & 31;
  const float* qr = Qt + q*128 + hh*32;
  const float* kr = Kh + (size_t)k*128 + hh*32;
  float acc = 0.f;
#pragma unroll
  for (int d = 0; d < 32; ++d) acc += qr[d] * kr[d];
  s[i] = acc * 0.17677669529663687f;
}

__global__ void softmax_rows(float* __restrict__ s, int len)
{
  __shared__ float red[8];
  float* r = s + (size_t)blockIdx.x * len;
  int tid = threadIdx.x;
  float m = -3.0e38f;
  for (int k = tid; k < len; k += blockDim.x) m = fmaxf(m, r[k]);
  m = wmaxv(m);
  if ((tid & 31) == 0) red[tid >> 5] = m;
  __syncthreads();
  float mm = red[0];
#pragma unroll
  for (int i = 1; i < 8; ++i) mm = fmaxf(mm, red[i]);
  __syncthreads();
  float z = 0.f;
  for (int k = tid; k < len; k += blockDim.x) { float e = expf(r[k] - mm); r[k] = e; z += e; }
  z = wsum(z);
  if ((tid & 31) == 0) red[tid >> 5] = z;
  __syncthreads();
  float zz = 0.f;
#pragma unroll
  for (int i = 0; i < 8; ++i) zz += red[i];
  float inv = 1.f / zz;
  for (int k = tid; k < len; k += blockDim.x) r[k] *= inv;
}

__global__ void attn_in_out(const float* __restrict__ a, const float* __restrict__ Vh,
                            float* __restrict__ o, int n)
{
  __shared__ float red[8];
  int bid = blockIdx.x;
  int d = bid & 31, q = (bid >> 5) & 31, hh = bid >> 10;
  const float* ar = a + (size_t)(hh*32 + q) * n;
  const float* vr = Vh + hh*32 + d;
  float acc = 0.f;
  for (int k = threadIdx.x; k < n; k += blockDim.x) acc += ar[k] * vr[(size_t)k * 128];
  acc = wsum(acc);
  if ((threadIdx.x & 31) == 0) red[threadIdx.x >> 5] = acc;
  __syncthreads();
  if (threadIdx.x == 0) {
    float t = 0.f;
#pragma unroll
    for (int i = 0; i < 8; ++i) t += red[i];
    o[q*128 + hh*32 + d] = t;
  }
}

// nodes attend to 32 tokens; K/V staged in LDS; one wave per node
__global__ void attn_out(const float* __restrict__ Qh, const float* __restrict__ Kt,
                         const float* __restrict__ Vt, float* __restrict__ out, int n)
{
  __shared__ float Ks[4096], Vs[4096];
  for (int i = threadIdx.x; i < 4096; i += blockDim.x) { Ks[i] = Kt[i]; Vs[i] = Vt[i]; }
  __syncthreads();
  int node = blockIdx.x * (blockDim.x >> 5) + (threadIdx.x >> 5);
  int lane = threadIdx.x & 31;
  if (node >= n) return;
  const float* qrow = Qh + (size_t)node * 128;
  float wv[4];
#pragma unroll
  for (int hh = 0; hh < 4; ++hh) {
    float acc = 0.f;
    const float* qp = qrow + hh*32;
    const float* kp = Ks + lane*128 + hh*32;     // token index = lane
#pragma unroll
    for (int d = 0; d < 32; ++d) acc += qp[d] * kp[d];
    acc *= 0.17677669529663687f;
    float m = wmaxv(acc);
    float e = expf(acc - m);
    float z = wsum(e);
    wv[hh] = e / z;
  }
#pragma unroll
  for (int hh = 0; hh < 4; ++hh) {
    float acc = 0.f;
    for (int t = 0; t < 32; ++t) {
      float wt = __shfl(wv[hh], t, 32);
      acc += wt * Vs[t*128 + hh*32 + lane];
    }
    out[(size_t)node*128 + hh*32 + lane] = acc;
  }
}

// ---------------- gather/scatter for the global sub-block ----------------
__global__ void gather_rows(const float* __restrict__ h, const int* __restrict__ idx,
                            float* __restrict__ o, int ng)
{
  int i = blockIdx.x * blockDim.x + threadIdx.x;
  if (i >= ng * 128) return;
  int j = i >> 7, d = i & 127;
  o[i] = h[(size_t)idx[j]*128 + d];
}
__global__ void scatter_rows(float* __restrict__ h, const int* __restrict__ idx,
                             const float* __restrict__ src, int ng)
{
  int i = blockIdx.x * blockDim.x + threadIdx.x;
  if (i >= ng * 128) return;
  int j = i >> 7, d = i & 127;
  h[(size_t)idx[j]*128 + d] = src[i];
}

// ---------------- host-side orchestration ----------------
static inline int cdiv(int a, int b){ return (a + b - 1) / b; }

struct MHAPk { const unsigned *wq,*wk,*wv,*wo; const float *bq,*bk,*bv,*bo; };
struct GFPk  { const unsigned *wl,*wr,*w1,*w2;
               const float *bl,*br,*att,*bias,*n1g,*n1b,*n2g,*n2b,*b1,*b2; };

static void run_gf(float* hbuf, int n, const int* esrc, const int* edst, int E,
                   const GFPk& q, float* xl, float* xr, float* gout, float* ybuf, float* ffn,
                   float* logitb, float* mb, float* zb, hipStream_t s)
{
  dim3 blk(128);
  gemm_wmma<0,128><<<dim3(8, cdiv(n,64)), blk, 0, s>>>(hbuf, q.wl, q.bl, nullptr, xl, n, 512);
  gemm_wmma<0,128><<<dim3(8, cdiv(n,64)), blk, 0, s>>>(hbuf, q.wr, q.br, nullptr, xr, n, 512);
  fillf<<<cdiv(n*4,256),256,0,s>>>(mb, -3.0e38f, (size_t)n*4);
  fillf<<<cdiv(n*4,256),256,0,s>>>(zb, 0.f, (size_t)n*4);
  fillf<<<cdiv(n*128,256),256,0,s>>>(gout, 0.f, (size_t)n*128);
  int tot = E + n;
  gat_logits <<<cdiv(tot,8),    256,0,s>>>(esrc, edst, E, n, xl, xr, q.att, logitb, mb);
  gat_expsum <<<cdiv(tot*4,256),256,0,s>>>(edst, E, n, logitb, mb, zb);
  gat_scatter<<<cdiv(tot,8),    256,0,s>>>(esrc, edst, E, n, xl, logitb, zb, gout);
  ln_rows<<<cdiv(n,8),256,0,s>>>(hbuf, gout, q.bias, q.n1g, q.n1b, ybuf, n);
  gemm_wmma<1,128><<<dim3(4, cdiv(n,64)), blk, 0, s>>>(ybuf, q.w1, q.b1, nullptr, ffn, n, 256);
  gemm_wmma<0,256><<<dim3(2, cdiv(n,64)), blk, 0, s>>>(ffn,  q.w2, q.b2, nullptr, gout, n, 128);
  ln_rows<<<cdiv(n,8),256,0,s>>>(ybuf, gout, nullptr, q.n2g, q.n2b, hbuf, n);
}

static void run_phys(float* h, int n, const float* tok0, const MHAPk& ain, const MHAPk& aout,
                     const float* n1g, const float* n1b, const float* n2g, const float* n2b,
                     float* Kh, float* Vh, float* sbuf, float* Qh,
                     float* attno, float* P, float* P2,
                     float* Qt, float* obuf, float* tok1, float* Kt, float* Vt, hipStream_t s)
{
  dim3 blk(128);
  dim3 g2(2, cdiv(n,64));
  gemm_wmma<0,128><<<dim3(2,1), blk,0,s>>>(tok0, ain.wq, ain.bq, nullptr, Qt, 32,128);
  gemm_wmma<0,128><<<g2, blk,0,s>>>(h, ain.wk, ain.bk, nullptr, Kh, n,128);
  gemm_wmma<0,128><<<g2, blk,0,s>>>(h, ain.wv, ain.bv, nullptr, Vh, n,128);
  attn_in_scores<<<cdiv(128*n,256),256,0,s>>>(Qt, Kh, sbuf, n);
  softmax_rows<<<128,256,0,s>>>(sbuf, n);
  attn_in_out<<<4096,256,0,s>>>(sbuf, Vh, obuf, n);
  gemm_wmma<0,128><<<dim3(2,1), blk,0,s>>>(obuf, ain.wo, ain.bo, tok0, tok1, 32,128);
  gemm_wmma<0,128><<<g2, blk,0,s>>>(h, aout.wq, aout.bq, nullptr, Qh, n,128);
  gemm_wmma<0,128><<<dim3(2,1), blk,0,s>>>(tok1, aout.wk, aout.bk, nullptr, Kt, 32,128);
  gemm_wmma<0,128><<<dim3(2,1), blk,0,s>>>(tok1, aout.wv, aout.bv, nullptr, Vt, 32,128);
  attn_out<<<cdiv(n,8),256,0,s>>>(Qh, Kt, Vt, attno, n);
  gemm_wmma<0,128><<<g2, blk,0,s>>>(attno, aout.wo, aout.bo, nullptr, P, n,128);
  ln_rows<<<cdiv(n,8),256,0,s>>>(P, nullptr, nullptr, n1g, n1b, P2, n);
  ln_rows<<<cdiv(n,8),256,0,s>>>(h, P2, nullptr, n2g, n2b, h, n);
}

extern "C" void kernel_launch(void* const* d_in, const int* in_sizes, int n_in,
                              void* d_out, int out_size, void* d_ws, size_t ws_size,
                              hipStream_t stream)
{
  (void)out_size; (void)ws_size;
  const float* x  = (const float*)d_in[0];
  const int* els  = (const int*)d_in[3];
  const int* eld  = (const int*)d_in[4];
  const int* idxg = (const int*)d_in[5];
  const int* egs  = (const int*)d_in[6];
  const int* egd  = (const int*)d_in[7];
  int N  = in_sizes[0] / 9;
  int EL = in_sizes[3];
  int NG = in_sizes[5];
  int EG = in_sizes[6];

  // --- parameter leaves (insertion-order flatten of the params dict) ---
  const float* pp[512]; int npar = 0;
  for (int i = 8; i < n_in && npar < 512; ++i) pp[npar++] = (const float*)d_in[i];
  int c = 0;
  auto nx = [&]() -> const float* { return pp[c++]; };
  const float *enc_w1=nx(),*enc_b1=nx(),*enc_w2=nx(),*enc_b2=nx();
  const float *dec_w1=nx(),*dec_b1=nx(),*dec_w2=nx(),*dec_b2=nx();
  struct LayerRaw {
    const float* tok;
    const float *ain[8], *aout[8];        // wq,bq,wk,bk,wv,bv,wo,bo
    const float *pn1g,*pn1b,*pn2g,*pn2b;
    const float *loc[14], *glob[14];      // wl,bl,wr,br,att,bias,n1g,n1b,n2g,n2b,w1,b1,w2,b2
  } LR[4];
  for (int l = 0; l < 4; ++l) {
    LR[l].tok = nx();
    for (int i = 0; i < 8; ++i) LR[l].ain[i]  = nx();
    for (int i = 0; i < 8; ++i) LR[l].aout[i] = nx();
    LR[l].pn1g = nx(); LR[l].pn1b = nx(); LR[l].pn2g = nx(); LR[l].pn2b = nx();
    for (int i = 0; i < 14; ++i) LR[l].loc[i]  = nx();
    for (int i = 0; i < 14; ++i) LR[l].glob[i] = nx();
  }

  // --- workspace carve ---
  char* w = (char*)d_ws; size_t off = 0;
  auto alloc = [&](size_t nf) -> float* {
    float* r = (float*)(w + off);
    off += ((nf * sizeof(float) + 255) & ~(size_t)255);
    return r;
  };
  float* h    = alloc((size_t)N * 128);
  float* xl   = alloc((size_t)N * 512);
  float* xr   = alloc((size_t)N * 512);
  float* ffn  = alloc((size_t)N * 256);
  float* gout = alloc((size_t)N * 128);
  float* ybuf = alloc((size_t)N * 128);
  float* hc   = alloc((size_t)N * 128);
  int Emax = (EL > EG ? EL : EG) + N;
  float* logitb = alloc((size_t)Emax * 4);
  float* mb = alloc((size_t)N * 4);
  float* zb = alloc((size_t)N * 4);
  // physics-phase aliases (xl/xr free during phys block)
  float* Kh    = xl;
  float* Vh    = xl + (size_t)N*128;
  float* sbuf  = xl + (size_t)N*256;
  float* Qh    = xl + (size_t)N*384;
  float* attno = xr;
  float* P     = xr + (size_t)N*128;
  float* P2    = xr + (size_t)N*256;
  float* sm    = xr + (size_t)N*384;
  float *Qt = sm, *obuf = sm + 4096, *tok1 = sm + 8192, *Kt = sm + 12288, *Vt = sm + 16384;

  // --- pre-pack every GEMM weight into f16 WMMA B-fragments ---
  auto packw = [&](const float* B, int K, int Nc) -> const unsigned* {
    unsigned* p = (unsigned*)alloc((size_t)K * Nc / 2);   // K*Nc/2 dwords
    int total = (Nc >> 4) * (K >> 5) * 256;
    pack_b_f16<<<cdiv(total,256),256,0,stream>>>(B, p, K, Nc);
    return p;
  };
  const unsigned* enc_w2p = packw(enc_w2, 128, 128);
  const unsigned* dec_w1p = packw(dec_w1, 128, 128);
  struct Layer { const float* tok; MHAPk ain, aout;
                 const float *pn1g,*pn1b,*pn2g,*pn2b; GFPk loc, glob; } L4[4];
  auto mkmha = [&](const float* const* r) -> MHAPk {
    MHAPk m;
    m.wq = packw(r[0],128,128); m.bq = r[1];
    m.wk = packw(r[2],128,128); m.bk = r[3];
    m.wv = packw(r[4],128,128); m.bv = r[5];
    m.wo = packw(r[6],128,128); m.bo = r[7];
    return m;
  };
  auto mkgf = [&](const float* const* r) -> GFPk {
    GFPk g;
    g.wl = packw(r[0],128,512); g.bl = r[1];
    g.wr = packw(r[2],128,512); g.br = r[3];
    g.att = r[4]; g.bias = r[5];
    g.n1g = r[6]; g.n1b = r[7]; g.n2g = r[8]; g.n2b = r[9];
    g.w1 = packw(r[10],128,256); g.b1 = r[11];
    g.w2 = packw(r[12],256,128); g.b2 = r[13];
    return g;
  };
  for (int l = 0; l < 4; ++l) {
    L4[l].tok  = LR[l].tok;
    L4[l].ain  = mkmha(LR[l].ain);
    L4[l].aout = mkmha(LR[l].aout);
    L4[l].pn1g = LR[l].pn1g; L4[l].pn1b = LR[l].pn1b;
    L4[l].pn2g = LR[l].pn2g; L4[l].pn2b = LR[l].pn2b;
    L4[l].loc  = mkgf(LR[l].loc);
    L4[l].glob = mkgf(LR[l].glob);
  }

  // --- encoder: h = gelu(x@w1+b1)@w2+b2 ---
  enc_first<<<cdiv(N*128,256),256,0,stream>>>(x, enc_w1, enc_b1, ybuf, N);
  gemm_wmma<0,128><<<dim3(2, cdiv(N,64)), dim3(128), 0, stream>>>(ybuf, enc_w2p, enc_b2,
                                                                  nullptr, h, N, 128);
  // HighFreqIndicator is computed-but-unused in the reference -> skipped.

  // --- layers ---
  for (int l = 0; l < 4; ++l) {
    Layer& Ly = L4[l];
    run_phys(h, N, Ly.tok, Ly.ain, Ly.aout, Ly.pn1g, Ly.pn1b, Ly.pn2g, Ly.pn2b,
             Kh, Vh, sbuf, Qh, attno, P, P2, Qt, obuf, tok1, Kt, Vt, stream);
    run_gf(h, N, els, eld, EL, Ly.loc, xl, xr, gout, ybuf, ffn, logitb, mb, zb, stream);
    // global block: only rows idx_global matter (self-loops close the dependence set)
    gather_rows<<<cdiv(NG*128,256),256,0,stream>>>(h, idxg, hc, NG);
    run_gf(hc, NG, egs, egd, EG, Ly.glob, xl, xr, gout, ybuf, ffn, logitb, mb, zb, stream);
    scatter_rows<<<cdiv(NG*128,256),256,0,stream>>>(h, idxg, hc, NG);
  }

  // --- decoder: out = gelu(h@w1+b1)@w2+b2 ---
  gemm_wmma<1,128><<<dim3(2, cdiv(N,64)), dim3(128), 0, stream>>>(h, dec_w1p, dec_b1,
                                                                  nullptr, ybuf, N, 128);
  dec_last<<<cdiv(N*4,256),256,0,stream>>>(ybuf, dec_w2, dec_b2, (float*)d_out, N);
}
